// ResponseDecoder_52828097741240
// MI455X (gfx1250) — matrix operand order repaired and checked by reading, not compiled, not run on hardware
//
#include <hip/hip_runtime.h>
#include <hip/hip_bf16.h>

// ---------------- problem constants ----------------
#define TT   128            // Tz == Tu
#define BB   64
#define HH   512
#define EE   512
#define VV   8000
#define DD   5
#define VT   (VV + TT)      // 8128
#define G3   (3 * HH)       // 1536
#define GRUIN (EE + 2*HH + DD) // 1541

typedef __attribute__((ext_vector_type(16))) __bf16 bf16x16;
typedef __attribute__((ext_vector_type(8)))  float  floatx8;
typedef __attribute__((ext_vector_type(4)))  float  floatx4;

// ---------------- fragment loaders (f32 -> bf16 on the fly) ----------------
// A 16x32 bf16 frag: lane (hf, m=lane&15), element j -> A[m][kbase + j]
// where kbase = k0 + 16*hf.  16 contiguous floats -> 4x float4.
__device__ __forceinline__ bf16x16 load_a_frag(const float* __restrict__ A,
                                               long row_off, int kbase) {
  const float* p = A + row_off + kbase;
  floatx4 f0 = *(const floatx4*)(p);
  floatx4 f1 = *(const floatx4*)(p + 4);
  floatx4 f2 = *(const floatx4*)(p + 8);
  floatx4 f3 = *(const floatx4*)(p + 12);
  bf16x16 r;
  r[0]=(__bf16)f0[0]; r[1]=(__bf16)f0[1]; r[2]=(__bf16)f0[2]; r[3]=(__bf16)f0[3];
  r[4]=(__bf16)f1[0]; r[5]=(__bf16)f1[1]; r[6]=(__bf16)f1[2]; r[7]=(__bf16)f1[3];
  r[8]=(__bf16)f2[0]; r[9]=(__bf16)f2[1]; r[10]=(__bf16)f2[2]; r[11]=(__bf16)f2[3];
  r[12]=(__bf16)f3[0]; r[13]=(__bf16)f3[1]; r[14]=(__bf16)f3[2]; r[15]=(__bf16)f3[3];
  return r;
}

// B 32x16 bf16 frag for C = A @ W^T: lane (hf, n=lane&15), element j -> W[n][kbase + j]
// kbase = k0 + 16*hf. 16 contiguous floats.
__device__ __forceinline__ bf16x16 load_b_frag_vec(const float* __restrict__ W,
                                                   long row_off, int kbase) {
  const float* p = W + row_off + kbase;
  floatx4 f0 = *(const floatx4*)(p);
  floatx4 f1 = *(const floatx4*)(p + 4);
  floatx4 f2 = *(const floatx4*)(p + 8);
  floatx4 f3 = *(const floatx4*)(p + 12);
  bf16x16 r;
  r[0]=(__bf16)f0[0]; r[1]=(__bf16)f0[1]; r[2]=(__bf16)f0[2]; r[3]=(__bf16)f0[3];
  r[4]=(__bf16)f1[0]; r[5]=(__bf16)f1[1]; r[6]=(__bf16)f1[2]; r[7]=(__bf16)f1[3];
  r[8]=(__bf16)f2[0]; r[9]=(__bf16)f2[1]; r[10]=(__bf16)f2[2]; r[11]=(__bf16)f2[3];
  r[12]=(__bf16)f3[0]; r[13]=(__bf16)f3[1]; r[14]=(__bf16)f3[2]; r[15]=(__bf16)f3[3];
  return r;
}

// scalar-load variant (W_ih has odd row stride 1541 -> float4 would be misaligned)
__device__ __forceinline__ bf16x16 load_b_frag_scalar(const float* __restrict__ W,
                                                      long row_off, int kbase) {
  const float* p = W + row_off + kbase;
  bf16x16 r;
  r[0]=(__bf16)p[0];  r[1]=(__bf16)p[1];  r[2]=(__bf16)p[2];  r[3]=(__bf16)p[3];
  r[4]=(__bf16)p[4];  r[5]=(__bf16)p[5];  r[6]=(__bf16)p[6];  r[7]=(__bf16)p[7];
  r[8]=(__bf16)p[8];  r[9]=(__bf16)p[9];  r[10]=(__bf16)p[10]; r[11]=(__bf16)p[11];
  r[12]=(__bf16)p[12]; r[13]=(__bf16)p[13]; r[14]=(__bf16)p[14]; r[15]=(__bf16)p[15];
  return r;
}

__device__ __forceinline__ float sigmoidf(float x) { return 1.0f / (1.0f + expf(-x)); }

// ---------------- block reductions ----------------
__device__ __forceinline__ float block_reduce_max(float v, float* sh) {
  #pragma unroll
  for (int off = 16; off > 0; off >>= 1) v = fmaxf(v, __shfl_xor(v, off, 32));
  int tid = threadIdx.x, nw = blockDim.x >> 5;
  if ((tid & 31) == 0) sh[tid >> 5] = v;
  __syncthreads();
  float r = sh[0];
  for (int i = 1; i < nw; ++i) r = fmaxf(r, sh[i]);
  __syncthreads();
  return r;
}
__device__ __forceinline__ float block_reduce_sum(float v, float* sh) {
  #pragma unroll
  for (int off = 16; off > 0; off >>= 1) v += __shfl_xor(v, off, 32);
  int tid = threadIdx.x, nw = blockDim.x >> 5;
  if ((tid & 31) == 0) sh[tid >> 5] = v;
  __syncthreads();
  float r = sh[0];
  for (int i = 1; i < nw; ++i) r += sh[i];
  __syncthreads();
  return r;
}

// ---------------- kernels ----------------

// C[m,n] = sum_k A[m,k]*W[n,k] + bias[n] (+ addend[m,n]).
// One wave per 16x32 output tile: one A frag feeds two independent WMMA chains.
template <int VECB>
__global__ void wmma_gemm_kernel(const float* __restrict__ A, int lda,
                                 const float* __restrict__ W, int ldw,
                                 const float* __restrict__ bias,
                                 const float* __restrict__ addend, int ldadd,
                                 float* __restrict__ C, int ldc,
                                 int M, int N, int K) {
  int warp = threadIdx.x >> 5;
  int tile = blockIdx.x * (blockDim.x >> 5) + warp;
  int tilesN2 = N >> 5;                         // 32-wide N tiles
  int total = (M >> 4) * tilesN2;
  if (tile >= total) return;                    // wave-uniform
  int m0 = (tile / tilesN2) << 4;
  int n0 = (tile % tilesN2) << 5;
  int lane = threadIdx.x & 31, hf = lane >> 4, ln = lane & 15;
  long arow  = (long)(m0 + ln) * lda;
  long brow0 = (long)(n0 + ln) * ldw;
  long brow1 = (long)(n0 + 16 + ln) * ldw;
  floatx8 acc0 = {0.f,0.f,0.f,0.f,0.f,0.f,0.f,0.f};
  floatx8 acc1 = {0.f,0.f,0.f,0.f,0.f,0.f,0.f,0.f};
  #pragma unroll 2
  for (int k0 = 0; k0 < K; k0 += 32) {
    __builtin_prefetch(W + brow0 + k0 + 64, 0, 3);   // speculative, near
    __builtin_prefetch(W + brow1 + k0 + 64, 0, 3);
    bf16x16 a = load_a_frag(A, arow, k0 + 16*hf);
    bf16x16 b0, b1;
    if (VECB) { b0 = load_b_frag_vec(W, brow0, k0 + 16*hf);
                b1 = load_b_frag_vec(W, brow1, k0 + 16*hf); }
    else      { b0 = load_b_frag_scalar(W, brow0, k0 + 16*hf);
                b1 = load_b_frag_scalar(W, brow1, k0 + 16*hf); }
    acc0 = __builtin_amdgcn_wmma_f32_16x16x32_bf16(false, a, false, b0,
                                                   (short)0, acc0, false, false);
    acc1 = __builtin_amdgcn_wmma_f32_16x16x32_bf16(false, a, false, b1,
                                                   (short)0, acc1, false, false);
  }
  int n_a = n0 + ln;
  int n_b = n0 + 16 + ln;
  float bv0 = bias ? bias[n_a] : 0.f;
  float bv1 = bias ? bias[n_b] : 0.f;
  #pragma unroll
  for (int i = 0; i < 8; ++i) {
    int m = m0 + i + (hf << 3);
    float v0 = acc0[i] + bv0;
    float v1 = acc1[i] + bv1;
    if (addend) { v0 += addend[(long)m * ldadd + n_a];
                  v1 += addend[(long)m * ldadd + n_b]; }
    C[(long)m * ldc + n_a] = v0;
    C[(long)m * ldc + n_b] = v1;
  }
}

// Fused energy: e[b,t] = sum_n red[n] * tanh( (W @ enc[t,b])[n] + rowAdd[b,n] + colAdd[n] )
// red[n] = vshared[n] or vrow[b,n].  enc rows are m = t*64+b; one wave per 16-row strip,
// loops all N in 32-wide steps (2 WMMA chains), reduces across lanes ->
// never materializes the (B*T, H) energy tensor.
__global__ void wmma_energy_kernel(const float* __restrict__ enc,
                                   const float* __restrict__ W, int ldw,
                                   const float* __restrict__ rowAdd,
                                   const float* __restrict__ colAdd,
                                   const float* __restrict__ vshared,
                                   const float* __restrict__ vrow,
                                   float* __restrict__ e_out, int T) {
  int warp = threadIdx.x >> 5;
  int strip = blockIdx.x * (blockDim.x >> 5) + warp;
  int m0 = strip << 4;
  if (m0 >= T * BB) return;                     // wave-uniform
  int lane = threadIdx.x & 31, hf = lane >> 4, ln = lane & 15;
  long arow = (long)(m0 + ln) * HH;
  float rs[8] = {0.f,0.f,0.f,0.f,0.f,0.f,0.f,0.f};
  for (int n0 = 0; n0 < HH; n0 += 32) {
    long brow0 = (long)(n0 + ln) * ldw;
    long brow1 = (long)(n0 + 16 + ln) * ldw;
    floatx8 acc0 = {0.f,0.f,0.f,0.f,0.f,0.f,0.f,0.f};
    floatx8 acc1 = {0.f,0.f,0.f,0.f,0.f,0.f,0.f,0.f};
    #pragma unroll 2
    for (int k0 = 0; k0 < HH; k0 += 32) {
      bf16x16 a  = load_a_frag(enc, arow, k0 + 16*hf);
      bf16x16 b0 = load_b_frag_vec(W, brow0, k0 + 16*hf);
      bf16x16 b1 = load_b_frag_vec(W, brow1, k0 + 16*hf);
      acc0 = __builtin_amdgcn_wmma_f32_16x16x32_bf16(false, a, false, b0,
                                                     (short)0, acc0, false, false);
      acc1 = __builtin_amdgcn_wmma_f32_16x16x32_bf16(false, a, false, b1,
                                                     (short)0, acc1, false, false);
    }
    int n_a = n0 + ln;
    int n_b = n0 + 16 + ln;
    float ca0 = colAdd ? colAdd[n_a] : 0.f;
    float ca1 = colAdd ? colAdd[n_b] : 0.f;
    #pragma unroll
    for (int i = 0; i < 8; ++i) {
      int r = m0 + i + (hf << 3);
      int bidx = r & (BB - 1);
      float pre0 = acc0[i] + ca0;
      float pre1 = acc1[i] + ca1;
      if (rowAdd) { pre0 += rowAdd[bidx * HH + n_a];
                    pre1 += rowAdd[bidx * HH + n_b]; }
      float wv0 = vshared ? vshared[n_a] : vrow[bidx * HH + n_a];
      float wv1 = vshared ? vshared[n_b] : vrow[bidx * HH + n_b];
      rs[i] += tanhf(pre0) * wv0 + tanhf(pre1) * wv1;
    }
  }
  #pragma unroll
  for (int off = 1; off < 16; off <<= 1) {
    #pragma unroll
    for (int i = 0; i < 8; ++i) rs[i] += __shfl_xor(rs[i], off, 32);
  }
  if (ln == 0) {                                // lanes 0 and 16 write 8 rows each
    #pragma unroll
    for (int i = 0; i < 8; ++i) {
      int r = m0 + i + (hf << 3);
      e_out[(r & (BB - 1)) * T + (r >> 6)] = rs[i];
    }
  }
}

__global__ void embed_gather_kernel(const float* __restrict__ emb,
                                    const int* __restrict__ mt,
                                    float* __restrict__ out) {
  int b = blockIdx.x;
  long row = (long)mt[b] * EE;
  for (int e = threadIdx.x; e < EE; e += blockDim.x)
    out[(long)b * EE + e] = emb[row + e];
}

// softmax over t + context:  ctx[b,h] = sum_t softmax(e[b,:])[t] * enc[t,b,h]
__global__ void softmax_ctx_kernel(const float* __restrict__ e,
                                   const float* __restrict__ enc,
                                   float* __restrict__ dst1, int ld1,
                                   float* __restrict__ dst2, int ld2, int T) {
  __shared__ float ash[TT];
  __shared__ float red[8];
  int b = blockIdx.x, tid = threadIdx.x;      // blockDim == T == 128
  float x = e[b * T + tid];
  float mx = block_reduce_max(x, red);
  float ex = expf(x - mx);
  ash[tid] = ex;
  float S = block_reduce_sum(ex, red);        // syncthreads inside publishes ash
  float inv = 1.0f / S;
  for (int h = tid; h < HH; h += blockDim.x) {
    float s = 0.f;
    for (int t = 0; t < T; ++t) s += ash[t] * enc[((long)t * BB + b) * HH + h];
    s *= inv;
    dst1[(long)b * ld1 + h] = s;
    dst2[(long)b * ld2 + h] = s;
  }
}

__global__ void gru_combine_kernel(const float* __restrict__ gi,
                                   const float* __restrict__ gh,
                                   const float* __restrict__ degree,
                                   const float* __restrict__ W_ih,
                                   const float* __restrict__ hprev,
                                   float* __restrict__ h_new,
                                   float* __restrict__ cat3,
                                   float* __restrict__ out_hidden,
                                   float* __restrict__ out_gru) {
  int idx = blockIdx.x * blockDim.x + threadIdx.x;
  if (idx >= BB * HH) return;
  int b = idx >> 9, h = idx & (HH - 1);
  float tr = 0.f, tz = 0.f, tn = 0.f;         // K=5 "degree" tail of gru_in
  #pragma unroll
  for (int d = 0; d < DD; ++d) {
    float dg = degree[b * DD + d];
    tr += dg * W_ih[(long)h            * GRUIN + G3 + d];
    tz += dg * W_ih[(long)(HH + h)     * GRUIN + G3 + d];
    tn += dg * W_ih[(long)(2*HH + h)   * GRUIN + G3 + d];
  }
  const float* gib = gi + (long)b * G3;
  const float* ghb = gh + (long)b * G3;
  float r  = sigmoidf(gib[h]        + tr + ghb[h]);
  float z  = sigmoidf(gib[HH + h]   + tz + ghb[HH + h]);
  float nn = tanhf  (gib[2*HH + h]  + tn + r * ghb[2*HH + h]);
  float hp = hprev[(long)b * HH + h];
  float hv = (1.0f - z) * nn + z * hp;
  h_new[(long)b * HH + h] = hv;
  cat3[(long)b * G3 + 2*HH + h] = hv;
  out_hidden[(long)b * HH + h] = hv;
  out_gru[(long)b * HH + h] = hv;
}

// sparse_z is (one-hot + 1e-10): log(sum_t e_t * sparse[t,b,:]) + m collapses to
// LDS scatter of 128 terms per batch row.
__global__ void sparse_copy_kernel(const float* __restrict__ zscore, // (B,T)
                                   const int* __restrict__ z_input,  // (T,B)
                                   const float* __restrict__ z_proba,// (T,B)
                                   float* __restrict__ zc) {         // (B,VT)
  __shared__ float ext[VT];
  __shared__ float red[8];
  int b = blockIdx.x, tid = threadIdx.x;      // blockDim == T == 128
  float s = zscore[b * TT + tid];
  float mx = block_reduce_max(s, red);
  float e = expf(s - mx);
  float S = block_reduce_sum(e, red);
  for (int v = tid; v < VT; v += blockDim.x) ext[v] = 0.f;
  __syncthreads();
  int zi = z_input[tid * BB + b];
  bool copy = (zi == 2) || (zi >= VV);
  if (copy)            atomicAdd(&ext[VV + tid], (5.0f - 1e-10f) * e);
  else if (zi != 0)    atomicAdd(&ext[zi], z_proba[tid * BB + b] * e);
  __syncthreads();
  for (int v = tid; v < VT; v += blockDim.x)
    zc[(long)b * VT + v] = logf(1e-10f * S + ext[v]) + mx;
}

// softmax over concat[gen(8000), zc(8128)] then merge copy-into-gen.
__global__ void final_softmax_kernel(const float* __restrict__ gen,
                                     const float* __restrict__ zc,
                                     float* __restrict__ proba) {
  __shared__ float red[8];
  int b = blockIdx.x, tid = threadIdx.x;      // blockDim = 256
  const float* gb = gen + (long)b * VV;
  const float* zb = zc + (long)b * VT;
  float mx = -3.402823466e38f;
  for (int v = tid; v < VV; v += blockDim.x) mx = fmaxf(mx, gb[v]);
  for (int v = tid; v < VT; v += blockDim.x) mx = fmaxf(mx, zb[v]);
  mx = block_reduce_max(mx, red);
  float s = 0.f;
  for (int v = tid; v < VV; v += blockDim.x) s += expf(gb[v] - mx);
  for (int v = tid; v < VT; v += blockDim.x) s += expf(zb[v] - mx);
  s = block_reduce_sum(s, red);
  float inv = 1.0f / s;
  float* pb = proba + (long)b * VT;
  for (int v = tid; v < VV; v += blockDim.x)
    pb[v] = (expf(gb[v] - mx) + expf(zb[v] - mx)) * inv;
  for (int v = VV + tid; v < VT; v += blockDim.x)
    pb[v] = expf(zb[v] - mx) * inv;
}

// ---------------- host-side launcher ----------------
extern "C" void kernel_launch(void* const* d_in, const int* in_sizes, int n_in,
                              void* d_out, int out_size, void* d_ws, size_t ws_size,
                              hipStream_t stream) {
  (void)in_sizes; (void)n_in; (void)out_size; (void)ws_size;
  const float* z_enc   = (const float*)d_in[0];
  const float* u_enc   = (const float*)d_in[1];
  const float* degree  = (const float*)d_in[2];
  const float* hidden  = (const float*)d_in[3];
  const float* z_proba = (const float*)d_in[4];
  const float* emb     = (const float*)d_in[5];
  const float* W_ctrl  = (const float*)d_in[6];
  const float* b_ctrl  = (const float*)d_in[7];
  const float* Wz      = (const float*)d_in[8];
  const float* bz      = (const float*)d_in[9];
  const float* vz      = (const float*)d_in[10];
  const float* Wu      = (const float*)d_in[11];
  const float* bu      = (const float*)d_in[12];
  const float* vu      = (const float*)d_in[13];
  const float* W_ih    = (const float*)d_in[14];
  const float* W_hh    = (const float*)d_in[15];
  const float* b_ih    = (const float*)d_in[16];
  const float* b_hh    = (const float*)d_in[17];
  const float* W_proj  = (const float*)d_in[18];
  const float* b_proj  = (const float*)d_in[19];
  const float* W_copy2 = (const float*)d_in[20];
  const float* b_copy2 = (const float*)d_in[21];
  const int*   m_t     = (const int*)d_in[22];
  const int*   z_in    = (const int*)d_in[23];

  float* out_proba  = (float*)d_out;                 // (64, 8128)
  float* out_hidden = out_proba + (long)BB * VT;     // (64, 512)
  float* out_gru    = out_hidden + (long)BB * HH;    // (64, 512)

  float* ws = (float*)d_ws;
  float* m_embed = ws;                     // 64*512
  float* x_cat   = m_embed + BB*EE;        // 64*1536  [m2 | u_ctx | z_ctx]
  float* cat3    = x_cat + BB*G3;          // 64*1536  [z_ctx | u_ctx | h_new]
  float* hz      = cat3 + BB*G3;           // 64*512
  float* hu      = hz + BB*HH;             // 64*512
  float* e_z     = hu + BB*HH;             // 64*128
  float* e_u     = e_z + BB*TT;            // 64*128
  float* gi      = e_u + BB*TT;            // 64*1536
  float* gh      = gi + BB*G3;             // 64*1536
  float* h_new   = gh + BB*G3;             // 64*512
  float* zscore  = h_new + BB*HH;          // 64*128
  float* zc      = zscore + BB*TT;         // 64*8128
  float* gen     = zc + (long)BB*VT;       // 64*8000

  // 1. m_embed = emb[m_t]
  embed_gather_kernel<<<BB, 256, 0, stream>>>(emb, m_t, m_embed);
  // 2. x_cat[:, :512] = m_embed + m_embed @ W_ctrl^T + b_ctrl   (64 tiles)
  wmma_gemm_kernel<1><<<16, 128, 0, stream>>>(m_embed, EE, W_ctrl, EE, b_ctrl,
                                              m_embed, EE, x_cat, G3, BB, EE, EE);
  // 3-4. hidden parts of both attentions: h @ W[:, :H]^T + b
  wmma_gemm_kernel<1><<<16, 128, 0, stream>>>(hidden, HH, Wz, 2*HH, bz,
                                              nullptr, 0, hz, HH, BB, HH, HH);
  wmma_gemm_kernel<1><<<16, 128, 0, stream>>>(hidden, HH, Wu, 2*HH, bu,
                                              nullptr, 0, hu, HH, BB, HH, HH);
  // 5-6. fused energies: e[b,t] = v . tanh(hpart[b] + W[:,H:] @ enc[t,b])
  wmma_energy_kernel<<<128, 128, 0, stream>>>(z_enc, Wz + HH, 2*HH, hz, nullptr,
                                              vz, nullptr, e_z, TT);
  wmma_energy_kernel<<<128, 128, 0, stream>>>(u_enc, Wu + HH, 2*HH, hu, nullptr,
                                              vu, nullptr, e_u, TT);
  // 7-8. softmax + context (write into gru-input slot and proj-input slot)
  softmax_ctx_kernel<<<BB, TT, 0, stream>>>(e_z, z_enc, x_cat + 2*HH, G3,
                                            cat3, G3, TT);
  softmax_ctx_kernel<<<BB, TT, 0, stream>>>(e_u, u_enc, x_cat + HH, G3,
                                            cat3 + HH, G3, TT);
  // 9. gh = hidden @ W_hh^T + b_hh   (192 tiles)
  wmma_gemm_kernel<1><<<48, 128, 0, stream>>>(hidden, HH, W_hh, HH, b_hh,
                                              nullptr, 0, gh, G3, BB, G3, HH);
  // 10. gi = x_cat @ W_ih[:, :1536]^T + b_ih  (odd ldw -> scalar B loads)
  wmma_gemm_kernel<0><<<48, 128, 0, stream>>>(x_cat, G3, W_ih, GRUIN, b_ih,
                                              nullptr, 0, gi, G3, BB, G3, G3);
  // 11. GRU gates (+ degree tail) -> h_new, cat3[:,1024:], outputs
  gru_combine_kernel<<<(BB*HH + 255)/256, 256, 0, stream>>>(
      gi, gh, degree, W_ih, hidden, h_new, cat3, out_hidden, out_gru);
  // 12. copy score: zscore[b,t] = h_new[b] . tanh(W_copy2 @ enc[t,b] + b_copy2)
  wmma_energy_kernel<<<128, 128, 0, stream>>>(z_enc, W_copy2, HH, nullptr,
                                              b_copy2, nullptr, h_new, zscore, TT);
  // 13. gen_score = cat3 @ W_proj^T + b_proj   (64 x 8000 x 1536, 1000 tiles)
  wmma_gemm_kernel<1><<<250, 128, 0, stream>>>(cat3, G3, W_proj, G3, b_proj,
                                               nullptr, 0, gen, VV, BB, VV, G3);
  // 14. sparse copy log-sum-exp scatter
  sparse_copy_kernel<<<BB, TT, 0, stream>>>(zscore, z_in, z_proba, zc);
  // 15. joint softmax + merge
  final_softmax_kernel<<<BB, 256, 0, stream>>>(gen, zc, out_proba);
}
